// TransformerBlock_14164802142807
// MI455X (gfx1250) — compile-verified, matrix operand order checked by reading, not run on hardware
//
#include <hip/hip_runtime.h>
#include <stdint.h>

typedef __bf16 bf16;
typedef bf16  v16bf __attribute__((ext_vector_type(16)));
typedef float v8f   __attribute__((ext_vector_type(8)));
typedef int   v4i   __attribute__((ext_vector_type(4)));

#define S_LEN     2048
#define D_MODEL   2048
#define NQ_H      32
#define NKV_H     8
#define HD_DIM    64
#define INTER_DIM 8192
#define QDIM      (NQ_H * HD_DIM)    // 2048
#define KVDIM     (NKV_H * HD_DIM)   // 512

__device__ __forceinline__ v8f wmma_bf16(v16bf a, v16bf b, v8f c) {
  return __builtin_amdgcn_wmma_f32_16x16x32_bf16(false, a, false, b, (short)0, c,
                                                 false, false);
}

// -------- async global->LDS staging (ASYNCcnt path), guarded with fallback ----
#if defined(__HIP_DEVICE_COMPILE__) && \
    __has_builtin(__builtin_amdgcn_global_load_async_to_lds_b128)
#define HAS_ASYNC_LDS 1
#else
#define HAS_ASYNC_LDS 0
#endif

#if HAS_ASYNC_LDS
typedef __attribute__((address_space(1))) v4i* gv4i_p;   // global int4*
typedef __attribute__((address_space(3))) v4i* lv4i_p;   // LDS int4*
#endif

__device__ __forceinline__ void copy16_g2l(const bf16* g, bf16* l) {
#if HAS_ASYNC_LDS
  __builtin_amdgcn_global_load_async_to_lds_b128((gv4i_p)(g), (lv4i_p)(l), 0, 0);
#else
  *reinterpret_cast<uint4*>(l) = *reinterpret_cast<const uint4*>(g);
#endif
}

__device__ __forceinline__ void async_copies_wait() {
#if HAS_ASYNC_LDS
#if __has_builtin(__builtin_amdgcn_s_wait_asynccnt)
  __builtin_amdgcn_s_wait_asynccnt(0);
#else
  asm volatile("s_wait_asynccnt 0" ::: "memory");
#endif
#endif
}

// ---------------------------------------------------------------- cvt f32->bf16
__global__ void cvt_bf16_kernel(const float* __restrict__ in, bf16* __restrict__ out,
                                size_t n) {
  size_t i = (size_t)blockIdx.x * blockDim.x + threadIdx.x;
  size_t stride = (size_t)gridDim.x * blockDim.x;
  for (; i < n; i += stride) out[i] = (bf16)in[i];
}

// ---------------------------------------------------------------- RMSNorm -> bf16
__global__ __launch_bounds__(256) void rmsnorm_bf16_kernel(
    const float* __restrict__ x, const float* __restrict__ g,
    bf16* __restrict__ out, int cols) {
  __shared__ float red[8];
  const int row = blockIdx.x;
  const int t = threadIdx.x;
  const float* xr = x + (size_t)row * cols;
  float ss = 0.f;
  for (int j = t; j < cols; j += 256) { float v = xr[j]; ss = fmaf(v, v, ss); }
#pragma unroll
  for (int off = 16; off > 0; off >>= 1) ss += __shfl_xor(ss, off, 32);
  if ((t & 31) == 0) red[t >> 5] = ss;
  __syncthreads();
  float tot = 0.f;
#pragma unroll
  for (int i = 0; i < 8; i++) tot += red[i];
  const float inv = rsqrtf(tot / (float)cols + 1e-5f);
  bf16* orow = out + (size_t)row * cols;
  for (int j = t; j < cols; j += 256) orow[j] = (bf16)(xr[j] * inv * g[j]);
}

// ---------------------------------------------------------------- RoPE -> bf16
__global__ void rope_bf16_kernel(const float* __restrict__ in, bf16* __restrict__ out,
                                 int nheads, int total) {
  int idx = blockIdx.x * 256 + threadIdx.x;
  if (idx >= total) return;
  int j = idx & 31;
  int head = (idx >> 5) % nheads;
  int s = idx / (32 * nheads);
  size_t base = (size_t)s * nheads * 64 + head * 64;
  float x1 = in[base + j], x2 = in[base + j + 32];
  float ang = (float)s * __powf(10000.f, -(float)j / 32.f);
  float sn, cs;
  sincosf(ang, &sn, &cs);
  out[base + j]      = (bf16)(x1 * cs - x2 * sn);
  out[base + j + 32] = (bf16)(x2 * cs + x1 * sn);
}

// ---------------------------------------------------------------- GEMM (bf16 WMMA)
// C[M,N] (f32) = A[M,K] @ B[K,N] (+ addend). 128x64x32 block, 8 waves,
// each wave owns a 32x32 tile (2x2 WMMA frags -> 4 wmma per K-step).
#define BM 128
#define BN 64
#define BK 32
#define LDA 40
#define LDB 40

__global__ __launch_bounds__(256) void gemm_bf16_kernel(
    const bf16* __restrict__ A, const bf16* __restrict__ B,
    float* __restrict__ C, const float* __restrict__ addend,
    int M, int N, int K) {
  __shared__ __align__(16) bf16 As[BM * LDA];
  __shared__ __align__(16) bf16 Bs[BN * LDB];
  const int t = threadIdx.x;
  const int m0 = blockIdx.y * BM, n0 = blockIdx.x * BN;
  const int wave = t >> 5, lane = t & 31;
  const int wm = wave >> 1, wn = wave & 1;   // wm 0..3 (32 rows), wn 0..1 (32 cols)
  const int r = lane & 15, hi = lane >> 4;
  const int arow = t >> 2, acol = (t & 3) * 8;   // A: 2 chunks of 64 rows x 32 cols
  const int brow = t >> 3, bcol = (t & 7) * 8;   // B: 32 rows x 64 cols

  v8f c00 = {}, c01 = {}, c10 = {}, c11 = {};

  for (int k0 = 0; k0 < K; k0 += BK) {
    // A tile: async global->LDS (byte-exact copy)
    copy16_g2l(A + (size_t)(m0 + arow) * K + k0 + acol, &As[arow * LDA + acol]);
    copy16_g2l(A + (size_t)(m0 + arow + 64) * K + k0 + acol,
               &As[(arow + 64) * LDA + acol]);
    // B tile: transpose-on-store to n-major so frag k-pairs are contiguous
    __align__(16) bf16 btmp[8];
    *reinterpret_cast<uint4*>(btmp) =
        *reinterpret_cast<const uint4*>(B + (size_t)(k0 + brow) * N + n0 + bcol);
#pragma unroll
    for (int i = 0; i < 8; i++) Bs[(bcol + i) * LDB + brow] = btmp[i];
    if (k0 + BK < K) {
      __builtin_prefetch(A + (size_t)(m0 + arow) * K + k0 + BK + acol, 0, 1);
      __builtin_prefetch(B + (size_t)(k0 + BK + brow) * N + n0 + bcol, 0, 1);
    }
    async_copies_wait();
    __syncthreads();

    v16bf a0, a1, b0, b1;
#pragma unroll
    for (int e = 0; e < 16; e++) {
      int kd = ((e & 8) ? 16 : 0) + hi * 8 + (e & 7);
      a0[e] = As[(wm * 32 + r) * LDA + kd];
      a1[e] = As[(wm * 32 + 16 + r) * LDA + kd];
      b0[e] = Bs[(wn * 32 + r) * LDB + kd];
      b1[e] = Bs[(wn * 32 + 16 + r) * LDB + kd];
    }
    c00 = wmma_bf16(a0, b0, c00);
    c01 = wmma_bf16(a0, b1, c01);
    c10 = wmma_bf16(a1, b0, c10);
    c11 = wmma_bf16(a1, b1, c11);
    __syncthreads();
  }

  v8f cf[4] = {c00, c01, c10, c11};
  if (addend) {
#pragma unroll
    for (int f = 0; f < 4; f++) {
      int nn = n0 + wn * 32 + (f & 1) * 16 + r;
#pragma unroll
      for (int i = 0; i < 8; i++) {
        int mm = m0 + wm * 32 + (f >> 1) * 16 + hi * 8 + i;
        C[(size_t)mm * N + nn] = cf[f][i] + addend[(size_t)mm * N + nn];
      }
    }
  } else {
#pragma unroll
    for (int f = 0; f < 4; f++) {
      int nn = n0 + wn * 32 + (f & 1) * 16 + r;
#pragma unroll
      for (int i = 0; i < 8; i++) {
        int mm = m0 + wm * 32 + (f >> 1) * 16 + hi * 8 + i;
        C[(size_t)mm * N + nn] = cf[f][i];
      }
    }
  }
}

// ---------------------------------------------------------------- dual GEMM + SwiGLU
// out[M,N] (bf16) = silu(A@B1) * (A@B2). Same tiling; 8 wmma per wave per K-step.
__global__ __launch_bounds__(256) void gemm_dual_swiglu_kernel(
    const bf16* __restrict__ A, const bf16* __restrict__ B1,
    const bf16* __restrict__ B2, bf16* __restrict__ out,
    int M, int N, int K) {
  __shared__ __align__(16) bf16 As[BM * LDA];
  __shared__ __align__(16) bf16 Bs1[BN * LDB];
  __shared__ __align__(16) bf16 Bs2[BN * LDB];
  const int t = threadIdx.x;
  const int m0 = blockIdx.y * BM, n0 = blockIdx.x * BN;
  const int wave = t >> 5, lane = t & 31;
  const int wm = wave >> 1, wn = wave & 1;
  const int r = lane & 15, hi = lane >> 4;
  const int arow = t >> 2, acol = (t & 3) * 8;
  const int brow = t >> 3, bcol = (t & 7) * 8;

  v8f u00 = {}, u01 = {}, u10 = {}, u11 = {};
  v8f w00 = {}, w01 = {}, w10 = {}, w11 = {};

  for (int k0 = 0; k0 < K; k0 += BK) {
    copy16_g2l(A + (size_t)(m0 + arow) * K + k0 + acol, &As[arow * LDA + acol]);
    copy16_g2l(A + (size_t)(m0 + arow + 64) * K + k0 + acol,
               &As[(arow + 64) * LDA + acol]);
    __align__(16) bf16 btmp[8];
    *reinterpret_cast<uint4*>(btmp) =
        *reinterpret_cast<const uint4*>(B1 + (size_t)(k0 + brow) * N + n0 + bcol);
#pragma unroll
    for (int i = 0; i < 8; i++) Bs1[(bcol + i) * LDB + brow] = btmp[i];
    *reinterpret_cast<uint4*>(btmp) =
        *reinterpret_cast<const uint4*>(B2 + (size_t)(k0 + brow) * N + n0 + bcol);
#pragma unroll
    for (int i = 0; i < 8; i++) Bs2[(bcol + i) * LDB + brow] = btmp[i];
    async_copies_wait();
    __syncthreads();

    v16bf a0, a1, p0, p1, q0v, q1v;
#pragma unroll
    for (int e = 0; e < 16; e++) {
      int kd = ((e & 8) ? 16 : 0) + hi * 8 + (e & 7);
      a0[e]  = As[(wm * 32 + r) * LDA + kd];
      a1[e]  = As[(wm * 32 + 16 + r) * LDA + kd];
      p0[e]  = Bs1[(wn * 32 + r) * LDB + kd];
      p1[e]  = Bs1[(wn * 32 + 16 + r) * LDB + kd];
      q0v[e] = Bs2[(wn * 32 + r) * LDB + kd];
      q1v[e] = Bs2[(wn * 32 + 16 + r) * LDB + kd];
    }
    u00 = wmma_bf16(a0, p0, u00);
    u01 = wmma_bf16(a0, p1, u01);
    u10 = wmma_bf16(a1, p0, u10);
    u11 = wmma_bf16(a1, p1, u11);
    w00 = wmma_bf16(a0, q0v, w00);
    w01 = wmma_bf16(a0, q1v, w01);
    w10 = wmma_bf16(a1, q0v, w10);
    w11 = wmma_bf16(a1, q1v, w11);
    __syncthreads();
  }

  v8f uf[4] = {u00, u01, u10, u11};
  v8f wf[4] = {w00, w01, w10, w11};
#pragma unroll
  for (int f = 0; f < 4; f++) {
    int nn = n0 + wn * 32 + (f & 1) * 16 + r;
#pragma unroll
    for (int i = 0; i < 8; i++) {
      int mm = m0 + wm * 32 + (f >> 1) * 16 + hi * 8 + i;
      float v1 = uf[f][i], v2 = wf[f][i];
      float sig = 1.f / (1.f + __expf(-v1));
      out[(size_t)mm * N + nn] = (bf16)(v1 * sig * v2);
    }
  }
}

// ---------------------------------------------------------------- flash attention
// One wave per (q-head, 16-row q tile). bf16 Q/K/V, fp32 online softmax, WMMA.
__global__ __launch_bounds__(32) void flash_attn_kernel(
    const bf16* __restrict__ Q,   // S x QDIM
    const bf16* __restrict__ Km,  // S x KVDIM
    const bf16* __restrict__ Vm,  // S x KVDIM
    float* __restrict__ O) {      // S x QDIM
  __shared__ __align__(4) bf16 ps[16 * 40];
  const int tiles = S_LEN >> 4;           // 128
  const int qtile = blockIdx.x % tiles;
  const int qhead = blockIdx.x / tiles;
  const int kvhead = qhead >> 2;          // G = NQ/NKV = 4
  const int lane = threadIdx.x;
  const int r = lane & 15, hi = lane >> 4;
  const int q0 = qtile * 16;

  v16bf aq0, aq1;
  const bf16* qrow = Q + (size_t)(q0 + r) * QDIM + qhead * HD_DIM;
#pragma unroll
  for (int e = 0; e < 16; e++) {
    int kd = ((e & 8) ? 16 : 0) + hi * 8 + (e & 7);
    aq0[e] = qrow[kd];
    aq1[e] = qrow[32 + kd];
  }

  float m_i[8], l_i[8];
  v8f oacc[4] = {};
#pragma unroll
  for (int i = 0; i < 8; i++) { m_i[i] = -1e30f; l_i[i] = 0.f; }

  const int nchunks = (q0 + 15) / 32 + 1;
  for (int jc = 0; jc < nchunks; ++jc) {
    const int kb = jc * 32;

    v8f s0 = {}, s1 = {};
    {
      const bf16* k0p = Km + (size_t)(kb + r) * KVDIM + kvhead * HD_DIM;
      const bf16* k1p = Km + (size_t)(kb + 16 + r) * KVDIM + kvhead * HD_DIM;
      v16bf bk00, bk01, bk10, bk11;
#pragma unroll
      for (int e = 0; e < 16; e++) {
        int kd = ((e & 8) ? 16 : 0) + hi * 8 + (e & 7);
        bk00[e] = k0p[kd];
        bk01[e] = k0p[32 + kd];
        bk10[e] = k1p[kd];
        bk11[e] = k1p[32 + kd];
      }
      s0 = wmma_bf16(aq0, bk00, s0);
      s0 = wmma_bf16(aq1, bk01, s0);
      s1 = wmma_bf16(aq0, bk10, s1);
      s1 = wmma_bf16(aq1, bk11, s1);
    }

#pragma unroll
    for (int i = 0; i < 8; i++) {
      int qr = q0 + hi * 8 + i;
      float x0 = (kb + r      <= qr) ? s0[i] * 0.125f : -3.0e38f;
      float x1 = (kb + 16 + r <= qr) ? s1[i] * 0.125f : -3.0e38f;
      float rm = fmaxf(x0, x1);
#pragma unroll
      for (int off = 1; off < 16; off <<= 1) rm = fmaxf(rm, __shfl_xor(rm, off, 32));
      float mn = fmaxf(m_i[i], rm);
      float al = __expf(m_i[i] - mn);
      float p0 = __expf(x0 - mn);
      float p1 = __expf(x1 - mn);
      float rs = p0 + p1;
#pragma unroll
      for (int off = 1; off < 16; off <<= 1) rs += __shfl_xor(rs, off, 32);
      l_i[i] = l_i[i] * al + rs;
      m_i[i] = mn;
#pragma unroll
      for (int tt = 0; tt < 4; tt++) oacc[tt][i] *= al;
      ps[(hi * 8 + i) * 40 + r]      = (bf16)p0;
      ps[(hi * 8 + i) * 40 + 16 + r] = (bf16)p1;
    }
    __syncthreads();   // single-wave block: orders LDS store->load

    v16bf ap;
#pragma unroll
    for (int e = 0; e < 16; e++) {
      int kk = ((e & 8) ? 16 : 0) + hi * 8 + (e & 7);
      ap[e] = ps[r * 40 + kk];
    }
    __syncthreads();

#pragma unroll
    for (int tt = 0; tt < 4; tt++) {
      v16bf bv;
#pragma unroll
      for (int e = 0; e < 16; e++) {
        int kvl = ((e & 8) ? 16 : 0) + hi * 8 + (e & 7);
        bv[e] = Vm[(size_t)(kb + kvl) * KVDIM + kvhead * HD_DIM + tt * 16 + r];
      }
      oacc[tt] = wmma_bf16(ap, bv, oacc[tt]);
    }
  }

#pragma unroll
  for (int tt = 0; tt < 4; tt++)
#pragma unroll
    for (int i = 0; i < 8; i++) {
      int mm = q0 + hi * 8 + i;
      O[(size_t)mm * QDIM + qhead * HD_DIM + tt * 16 + r] = oacc[tt][i] / l_i[i];
    }
}

// ---------------------------------------------------------------- launch
extern "C" void kernel_launch(void* const* d_in, const int* in_sizes, int n_in,
                              void* d_out, int out_size, void* d_ws, size_t ws_size,
                              hipStream_t stream) {
  const float* x  = (const float*)d_in[0];
  const float* Wq = (const float*)d_in[1];
  const float* Wk = (const float*)d_in[2];
  const float* Wv = (const float*)d_in[3];
  const float* Wo = (const float*)d_in[4];
  const float* W1 = (const float*)d_in[5];
  const float* W2 = (const float*)d_in[6];
  const float* W3 = (const float*)d_in[7];
  const float* g1 = (const float*)d_in[8];
  const float* g2 = (const float*)d_in[9];
  float* out = (float*)d_out;

  char* ws = (char*)d_ws;
  size_t off = 0;
  auto alloc = [&](size_t bytes) -> char* {
    char* p = ws + off;
    off += (bytes + 255) & ~(size_t)255;
    return p;
  };

  bf16* h_bf  = (bf16*)alloc((size_t)S_LEN * D_MODEL * 2);
  bf16* wq_bf = (bf16*)alloc((size_t)D_MODEL * QDIM * 2);
  bf16* wk_bf = (bf16*)alloc((size_t)D_MODEL * KVDIM * 2);
  bf16* wv_bf = (bf16*)alloc((size_t)D_MODEL * KVDIM * 2);
  bf16* wo_bf = (bf16*)alloc((size_t)D_MODEL * D_MODEL * 2);
  bf16* w1_bf = (bf16*)alloc((size_t)D_MODEL * INTER_DIM * 2);
  bf16* w2_bf = (bf16*)alloc((size_t)D_MODEL * INTER_DIM * 2);
  bf16* w3_bf = (bf16*)alloc((size_t)INTER_DIM * D_MODEL * 2);
  float* qf   = (float*)alloc((size_t)S_LEN * QDIM * 4);
  float* kf   = (float*)alloc((size_t)S_LEN * KVDIM * 4);
  float* vf   = (float*)alloc((size_t)S_LEN * KVDIM * 4);
  bf16* qb    = (bf16*)alloc((size_t)S_LEN * QDIM * 2);
  bf16* kb    = (bf16*)alloc((size_t)S_LEN * KVDIM * 2);
  bf16* vb    = (bf16*)alloc((size_t)S_LEN * KVDIM * 2);
  float* attnf = (float*)alloc((size_t)S_LEN * QDIM * 4);
  bf16* attnb  = (bf16*)alloc((size_t)S_LEN * QDIM * 2);
  float* xres  = (float*)alloc((size_t)S_LEN * D_MODEL * 4);
  bf16* h2b    = (bf16*)alloc((size_t)S_LEN * D_MODEL * 2);
  bf16* mlpb   = (bf16*)alloc((size_t)S_LEN * INTER_DIM * 2);

  cvt_bf16_kernel<<<4096, 256, 0, stream>>>(Wq, wq_bf, (size_t)D_MODEL * QDIM);
  cvt_bf16_kernel<<<4096, 256, 0, stream>>>(Wk, wk_bf, (size_t)D_MODEL * KVDIM);
  cvt_bf16_kernel<<<4096, 256, 0, stream>>>(Wv, wv_bf, (size_t)D_MODEL * KVDIM);
  cvt_bf16_kernel<<<4096, 256, 0, stream>>>(Wo, wo_bf, (size_t)D_MODEL * D_MODEL);
  cvt_bf16_kernel<<<4096, 256, 0, stream>>>(W1, w1_bf, (size_t)D_MODEL * INTER_DIM);
  cvt_bf16_kernel<<<4096, 256, 0, stream>>>(W2, w2_bf, (size_t)D_MODEL * INTER_DIM);
  cvt_bf16_kernel<<<4096, 256, 0, stream>>>(W3, w3_bf, (size_t)INTER_DIM * D_MODEL);

  rmsnorm_bf16_kernel<<<S_LEN, 256, 0, stream>>>(x, g1, h_bf, D_MODEL);

  gemm_bf16_kernel<<<dim3(QDIM / BN, S_LEN / BM), 256, 0, stream>>>(
      h_bf, wq_bf, qf, nullptr, S_LEN, QDIM, D_MODEL);
  gemm_bf16_kernel<<<dim3(KVDIM / BN, S_LEN / BM), 256, 0, stream>>>(
      h_bf, wk_bf, kf, nullptr, S_LEN, KVDIM, D_MODEL);
  gemm_bf16_kernel<<<dim3(KVDIM / BN, S_LEN / BM), 256, 0, stream>>>(
      h_bf, wv_bf, vf, nullptr, S_LEN, KVDIM, D_MODEL);

  {
    int totq = S_LEN * NQ_H * 32;
    rope_bf16_kernel<<<(totq + 255) / 256, 256, 0, stream>>>(qf, qb, NQ_H, totq);
    int totk = S_LEN * NKV_H * 32;
    rope_bf16_kernel<<<(totk + 255) / 256, 256, 0, stream>>>(kf, kb, NKV_H, totk);
  }
  cvt_bf16_kernel<<<2048, 256, 0, stream>>>(vf, vb, (size_t)S_LEN * KVDIM);

  flash_attn_kernel<<<NQ_H * (S_LEN / 16), 32, 0, stream>>>(qb, kb, vb, attnf);
  cvt_bf16_kernel<<<4096, 256, 0, stream>>>(attnf, attnb, (size_t)S_LEN * QDIM);

  gemm_bf16_kernel<<<dim3(D_MODEL / BN, S_LEN / BM), 256, 0, stream>>>(
      attnb, wo_bf, xres, x, S_LEN, D_MODEL, D_MODEL);

  rmsnorm_bf16_kernel<<<S_LEN, 256, 0, stream>>>(xres, g2, h2b, D_MODEL);

  gemm_dual_swiglu_kernel<<<dim3(INTER_DIM / BN, S_LEN / BM), 256, 0, stream>>>(
      h2b, w1_bf, w2_bf, mlpb, S_LEN, INTER_DIM, D_MODEL);

  gemm_bf16_kernel<<<dim3(D_MODEL / BN, S_LEN / BM), 256, 0, stream>>>(
      mlpb, w3_bf, out, xres, S_LEN, D_MODEL, INTER_DIM);

  (void)in_sizes; (void)n_in; (void)out_size; (void)ws_size;
}
